// MultiHeadAttention_103079215839
// MI455X (gfx1250) — matrix operand order repaired and checked by reading, not compiled, not run on hardware
//
#include <hip/hip_runtime.h>
#include <hip/hip_bf16.h>

// MI455X (gfx1250) wave32 WMMA implementation of MHA forward.
// B=2, S=2048, E=1024, H=16, D=64.  SCALE = 1/8.
//
// ws layout (f16): q[B,H,S,D] | k[B,H,S,D] | v[B,H,D,S] | o[B,S,E] | Wq16|Wk16|Wv16|Wo16
//                  4 x 8MB                                          + 4 x 2MB  = 40MB

#define Bb 2
#define Ss 2048
#define Ee 1024
#define Hh 16
#define Dd 64

typedef __attribute__((ext_vector_type(16))) _Float16 v16h;
typedef __attribute__((ext_vector_type(8)))  _Float16 v8h;
typedef __attribute__((ext_vector_type(4)))  _Float16 v4h;
typedef __attribute__((ext_vector_type(8)))  float    v8f;

static __device__ __forceinline__ v16h cat16(v8h lo, v8h hi) {
  return __builtin_shufflevector(lo, hi, 0,1,2,3,4,5,6,7,8,9,10,11,12,13,14,15);
}

static __device__ __forceinline__ v8f wmma_f16(v16h a, v16h b, v8f c) {
  // D = A(16x32 f16) x B(32x16 f16) + C(16x16 f32)
  return __builtin_amdgcn_wmma_f32_16x16x32_f16(false, a, false, b, (short)0, c, false, false);
}

static __device__ __forceinline__ v16h cvt16(float4 a, float4 b, float4 c, float4 d) {
  v16h r = { (_Float16)a.x,(_Float16)a.y,(_Float16)a.z,(_Float16)a.w,
             (_Float16)b.x,(_Float16)b.y,(_Float16)b.z,(_Float16)b.w,
             (_Float16)c.x,(_Float16)c.y,(_Float16)c.z,(_Float16)c.w,
             (_Float16)d.x,(_Float16)d.y,(_Float16)d.z,(_Float16)d.w };
  return r;
}

// ---------------------------------------------------------------------------
// f32 -> f16 weight pre-conversion (weights are reused by 256 M-tiles; convert
// once instead of per-tile in the GEMM hot loop).  n4 = elements/4.
// ---------------------------------------------------------------------------
__global__ void cvt_f32_to_f16(const float* __restrict__ src,
                               _Float16* __restrict__ dst, int n4) {
  const int i = blockIdx.x * blockDim.x + threadIdx.x;
  if (i < n4) {
    const float4 f = ((const float4*)src)[i];
    v4h h = { (_Float16)f.x, (_Float16)f.y, (_Float16)f.z, (_Float16)f.w };
    ((v4h*)dst)[i] = h;
  }
}

// ---------------------------------------------------------------------------
// GEMM: Out[m,n] = sum_k X[m,k]*W16[n,k] + bias[n],  X:[4096,1024], W16 f16.
// One wave computes a 16(M) x 64(N) tile.  4096 waves total.
// XF16: X is _Float16 (ws) else float (global input, converted on the fly).
// OUT_MODE: 0 -> f16 [B,H,S,D]; 1 -> f16 [B,H,D,S]; 2 -> f32 row-major [m,n]
// ---------------------------------------------------------------------------
template <bool XF16, int OUT_MODE>
__global__ void proj_gemm(const void* __restrict__ Xv,
                          const _Float16* __restrict__ W16,
                          const float* __restrict__ bias,
                          void* __restrict__ Outv) {
  const int lane = threadIdx.x & 31;
  const int wave = threadIdx.x >> 5;
  const int wid  = blockIdx.x * (blockDim.x >> 5) + wave;   // 0..4095
  const int m0   = (wid >> 4) << 4;                          // 256 M-tiles
  const int n0   = (wid & 15) << 6;                          // 16  N-tiles of 64
  const int half = lane >> 4;                                // 0: lanes 0-15, 1: 16-31
  const int l16  = lane & 15;

  const int mrow = m0 + l16;          // A row for this lane
  const int alo  = half * 8;          // A K-split per ISA layout
  const int boff = half * 16;         // B K-half per lane group

  v8f acc[4];
#pragma unroll
  for (int j = 0; j < 4; ++j)
#pragma unroll
    for (int r = 0; r < 8; ++r) acc[j][r] = 0.0f;

  for (int k0 = 0; k0 < Ee; k0 += 32) {
    v16h a;
    if (XF16) {
      const _Float16* X = (const _Float16*)Xv;
      const _Float16* xp = X + (size_t)mrow * Ee + k0 + alo;
      a = cat16(*(const v8h*)xp, *(const v8h*)(xp + 16));
    } else {
      const float* X = (const float*)Xv;
      const float4* p0 = (const float4*)(X + (size_t)mrow * Ee + k0 + alo);
      const float4* p1 = (const float4*)(X + (size_t)mrow * Ee + k0 + alo + 16);
      a = cvt16(p0[0], p0[1], p1[0], p1[1]);
    }
#pragma unroll
    for (int j = 0; j < 4; ++j) {
      const int nrow = n0 + 16 * j + l16;
      const _Float16* wp = W16 + (size_t)nrow * Ee + k0 + alo;    // two 8-f16 chunks (k 8h.., 16+8h..)
      v16h bf = cat16(*(const v8h*)wp, *(const v8h*)(wp + 16));
      acc[j] = wmma_f16(a, bf, acc[j]);
    }
  }

#pragma unroll
  for (int j = 0; j < 4; ++j) {
    const int n  = n0 + 16 * j + l16;
    const float bn = bias[n];
#pragma unroll
    for (int r = 0; r < 8; ++r) {
      const int m = m0 + r + half * 8;
      const float val = acc[j][r] + bn;
      if (OUT_MODE == 2) {
        ((float*)Outv)[(size_t)m * Ee + n] = val;
      } else {
        const int b = m >> 11, s = m & (Ss - 1);
        const int h = n >> 6,  d = n & (Dd - 1);
        const _Float16 hv = (_Float16)val;
        if (OUT_MODE == 0)
          ((_Float16*)Outv)[(((size_t)b * Hh + h) * Ss + s) * Dd + d] = hv;
        else
          ((_Float16*)Outv)[(((size_t)b * Hh + h) * Dd + d) * Ss + s] = hv;
      }
    }
  }
}

// ---------------------------------------------------------------------------
// Flash attention: one wave per (b,h, 16-row q-tile). 4096 waves.
// Streams 32 keys per iteration: 4 WMMA (QK^T) + online softmax + 4 WMMA (PV).
// rel_bias streamed with non-temporal loads (512MB, read once -> HBM-bound).
// ---------------------------------------------------------------------------
__global__ void attn_flash(const _Float16* __restrict__ qw,
                           const _Float16* __restrict__ kw,
                           const _Float16* __restrict__ vw,
                           const float* __restrict__ rel_bias,
                           _Float16* __restrict__ ow) {
  __shared__ alignas(16) _Float16 lds[4][16 * 32];   // P-tile transpose staging, per wave

  const int lane  = threadIdx.x & 31;
  const int wave  = threadIdx.x >> 5;
  const int wid   = blockIdx.x * 4 + wave;     // 0..4095
  const int qtile = wid & 127;
  const int bh    = wid >> 7;                  // b*H + h
  const int m0    = qtile * 16;
  const int half  = lane >> 4;
  const int l16   = lane & 15;

  const _Float16* qbase = qw + (size_t)bh * Ss * Dd;
  const _Float16* kbase = kw + (size_t)bh * Ss * Dd;
  const _Float16* vbase = vw + (size_t)bh * Dd * Ss;       // [D,S]
  const float*    bbase = rel_bias + (size_t)bh * Ss * Ss;
  _Float16*       myl   = lds[wave];

  // Q A-fragments (resident): K-dim chunks c=0 (d 0..31), c=1 (d 32..63)
  const int arow = m0 + l16;
  const int alo  = half * 8;
  v16h qa[2];
#pragma unroll
  for (int c = 0; c < 2; ++c) {
    const _Float16* qp = qbase + (size_t)arow * Dd + 32 * c + alo;
    qa[c] = cat16(*(const v8h*)qp, *(const v8h*)(qp + 16));
  }

  float m_r[8], l_r[8];
  v8f o[4];
#pragma unroll
  for (int r = 0; r < 8; ++r) { m_r[r] = -3.0e38f; l_r[r] = 0.0f; }
#pragma unroll
  for (int f = 0; f < 4; ++f)
#pragma unroll
    for (int r = 0; r < 8; ++r) o[f][r] = 0.0f;

  for (int j0 = 0; j0 < Ss; j0 += 32) {
    // ---- scores: S0 = keys j0..j0+15, S1 = keys j0+16..j0+31 ----
    v8f s0, s1;
#pragma unroll
    for (int r = 0; r < 8; ++r) { s0[r] = 0.0f; s1[r] = 0.0f; }
#pragma unroll
    for (int c = 0; c < 2; ++c) {
      const _Float16* kp0 = kbase + (size_t)(j0 + l16) * Dd + 32 * c + half * 8;
      const _Float16* kp1 = kbase + (size_t)(j0 + 16 + l16) * Dd + 32 * c + half * 8;
      v16h kb0 = cat16(*(const v8h*)kp0, *(const v8h*)(kp0 + 16));
      v16h kb1 = cat16(*(const v8h*)kp1, *(const v8h*)(kp1 + 16));
      s0 = wmma_f16(qa[c], kb0, s0);
      s1 = wmma_f16(qa[c], kb1, s1);
    }

    // ---- scale + rel_bias (streamed NT) + online softmax ----
    float p0[8], p1[8];
#pragma unroll
    for (int r = 0; r < 8; ++r) {
      const size_t q = (size_t)(m0 + r + half * 8);
      const float* bp = bbase + q * Ss + j0 + l16;
      p0[r] = s0[r] * 0.125f + __builtin_nontemporal_load(bp);
      p1[r] = s1[r] * 0.125f + __builtin_nontemporal_load(bp + 16);
    }
#pragma unroll
    for (int r = 0; r < 8; ++r) {
      float mx = fmaxf(p0[r], p1[r]);
      mx = fmaxf(mx, __shfl_xor(mx, 1));
      mx = fmaxf(mx, __shfl_xor(mx, 2));
      mx = fmaxf(mx, __shfl_xor(mx, 4));
      mx = fmaxf(mx, __shfl_xor(mx, 8));          // row-max within 16-lane half
      const float mi    = fmaxf(m_r[r], mx);
      const float alpha = __expf(m_r[r] - mi);
      m_r[r] = mi;
      p0[r] = __expf(p0[r] - mi);
      p1[r] = __expf(p1[r] - mi);
      float rs = p0[r] + p1[r];
      rs += __shfl_xor(rs, 1);
      rs += __shfl_xor(rs, 2);
      rs += __shfl_xor(rs, 4);
      rs += __shfl_xor(rs, 8);                    // row-sum
      l_r[r] = l_r[r] * alpha + rs;
#pragma unroll
      for (int f = 0; f < 4; ++f) o[f][r] *= alpha;
    }

    // ---- P: C-layout -> A-layout via wave-private LDS tile ----
#pragma unroll
    for (int r = 0; r < 8; ++r) {
      const int row = r + half * 8;
      myl[row * 32 + l16]      = (_Float16)p0[r];
      myl[row * 32 + 16 + l16] = (_Float16)p1[r];
    }
    // LDS ops of one wave execute in order; barrier the compiler + drain DScnt.
    asm volatile("s_wait_dscnt 0" ::: "memory");

    const _Float16* pr = myl + l16 * 32 + half * 8;
    v16h pa = cat16(*(const v8h*)pr, *(const v8h*)(pr + 16));

    // ---- O += P @ V  (V stored [D,S] -> contiguous B-fragments) ----
#pragma unroll
    for (int f = 0; f < 4; ++f) {
      const _Float16* vp = vbase + (size_t)(f * 16 + l16) * Ss + j0 + half * 8;
      v16h vb = cat16(*(const v8h*)vp, *(const v8h*)(vp + 16));
      o[f] = wmma_f16(pa, vb, o[f]);
    }
  }

  // ---- finalize: O /= l, store f16 [B,S,E] for the output projection ----
  const int b = bh >> 4, h = bh & 15;
#pragma unroll
  for (int r = 0; r < 8; ++r) {
    const float inv = 1.0f / l_r[r];
    const int s = m0 + r + half * 8;
#pragma unroll
    for (int f = 0; f < 4; ++f) {
      const int n = h * Dd + f * 16 + l16;
      ow[((size_t)b * Ss + s) * Ee + n] = (_Float16)(o[f][r] * inv);
    }
  }
}

// ---------------------------------------------------------------------------
extern "C" void kernel_launch(void* const* d_in, const int* in_sizes, int n_in,
                              void* d_out, int out_size, void* d_ws, size_t ws_size,
                              hipStream_t stream) {
  const float* query = (const float*)d_in[0];
  const float* key   = (const float*)d_in[1];
  const float* value = (const float*)d_in[2];
  const float* Wq    = (const float*)d_in[3];
  const float* bq    = (const float*)d_in[4];
  const float* Wk    = (const float*)d_in[5];
  const float* bk    = (const float*)d_in[6];
  const float* Wv    = (const float*)d_in[7];
  const float* bv    = (const float*)d_in[8];
  const float* Wo    = (const float*)d_in[9];
  const float* bo    = (const float*)d_in[10];
  const float* relb  = (const float*)d_in[11];

  const size_t QKV = (size_t)Bb * Hh * Ss * Dd;   // 4M f16 elems each
  const size_t WSZ = (size_t)Ee * Ee;             // 1M f16 elems each
  _Float16* qws  = (_Float16*)d_ws;
  _Float16* kws  = qws + QKV;
  _Float16* vws  = kws + QKV;
  _Float16* ows  = vws + QKV;
  _Float16* wq16 = ows + QKV;
  _Float16* wk16 = wq16 + WSZ;
  _Float16* wv16 = wk16 + WSZ;
  _Float16* wo16 = wv16 + WSZ;                    // total 40MB of ws

  const dim3 grid(1024), block(128);              // 4 waves/block, 4096 waves
  const int  n4 = (int)(WSZ / 4);                 // 262144 float4 per weight

  cvt_f32_to_f16<<<dim3((n4 + 255) / 256), dim3(256), 0, stream>>>(Wq, wq16, n4);
  cvt_f32_to_f16<<<dim3((n4 + 255) / 256), dim3(256), 0, stream>>>(Wk, wk16, n4);
  cvt_f32_to_f16<<<dim3((n4 + 255) / 256), dim3(256), 0, stream>>>(Wv, wv16, n4);
  cvt_f32_to_f16<<<dim3((n4 + 255) / 256), dim3(256), 0, stream>>>(Wo, wo16, n4);

  proj_gemm<false, 0><<<grid, block, 0, stream>>>(query, wq16, bq, qws);
  proj_gemm<false, 0><<<grid, block, 0, stream>>>(key,   wk16, bk, kws);
  proj_gemm<false, 1><<<grid, block, 0, stream>>>(value, wv16, bv, vws);
  attn_flash<<<grid, block, 0, stream>>>(qws, kws, vws, relb, ows);
  proj_gemm<true, 2><<<grid, block, 0, stream>>>(ows, wo16, bo, (float*)d_out);
}
